// SlidingWindowAttention_72524817760455
// MI455X (gfx1250) — matrix operand order repaired and checked by reading, not compiled, not run on hardware
//
#include <hip/hip_runtime.h>
#include <hip/hip_bf16.h>

// ---------------- constants ----------------
#define S_LEN   4096
#define HIDDEN  1024
#define HEADS   16
#define HDIM    64
#define WIN     128      // attends |i-j| <= 128

typedef __bf16 bf16;
typedef bf16  v16bf __attribute__((ext_vector_type(16)));
typedef bf16  v8bf  __attribute__((ext_vector_type(8)));
typedef float v8f   __attribute__((ext_vector_type(8)));

typedef unsigned int u32x4 __attribute__((ext_vector_type(4)));
typedef int          i32x8 __attribute__((ext_vector_type(8)));
typedef int          i32x4 __attribute__((ext_vector_type(4)));

#if defined(__has_builtin)
#  if __has_builtin(__builtin_amdgcn_tensor_load_to_lds)
#    define HAVE_TDM 1
#  endif
#endif

// ---------------- helpers ----------------
static __device__ __forceinline__ bf16 f2bf(float f) {
    union { float f; unsigned u; } x; x.f = f;
    unsigned r = x.u + 0x7FFFu + ((x.u >> 16) & 1u);   // round-to-nearest-even
    unsigned short h = (unsigned short)(r >> 16);
    bf16 b;
    __builtin_memcpy(&b, &h, 2);
    return b;
}

static __device__ __forceinline__ v16bf cat8(v8bf lo, v8bf hi) {
    return __builtin_shufflevector(lo, hi, 0,1,2,3,4,5,6,7,8,9,10,11,12,13,14,15);
}

static __device__ __forceinline__ float redmax16(float v) {
    v = fmaxf(v, __shfl_xor(v, 1, 32));
    v = fmaxf(v, __shfl_xor(v, 2, 32));
    v = fmaxf(v, __shfl_xor(v, 4, 32));
    v = fmaxf(v, __shfl_xor(v, 8, 32));
    return v;
}
static __device__ __forceinline__ float redsum16(float v) {
    v += __shfl_xor(v, 1, 32);
    v += __shfl_xor(v, 2, 32);
    v += __shfl_xor(v, 4, 32);
    v += __shfl_xor(v, 8, 32);
    return v;
}

#ifdef HAVE_TDM
// Issue one Tensor-Data-Mover 2D tile load: tile0 x tile1 elements (2B each)
// from a row-major tensor [tdim1][tdim0] (row stride = stride0 elements)
// starting at gaddr, into LDS at byte offset lds_off. D# layout per ISA 8.3/8.4.
// 6-arg toolchain form: (g0, g1, g2, g3, g_ext, cpol); groups 2/3 zero => 2D.
static __device__ __forceinline__ void tdm_load_tile_2d(const bf16* gaddr,
                                                        unsigned lds_off,
                                                        unsigned tdim0, unsigned tdim1,
                                                        unsigned stride0,
                                                        unsigned tile0, unsigned tile1) {
    unsigned long long ga = (unsigned long long)(uintptr_t)gaddr;
    u32x4 g0;
    g0[0] = 1u;                                             // count=1, user descriptor
    g0[1] = lds_off;                                        // lds_addr (bytes)
    g0[2] = (unsigned)ga;                                   // global_addr[31:0]
    g0[3] = (unsigned)((ga >> 32) & 0x1FFFFFFu) | (2u << 30); // addr[56:32] | type=2
    i32x8 g1;
    g1[0] = (int)(1u << 16);                                // data_size=1 (2 bytes), mask=0
    g1[1] = (int)((tdim0 & 0xFFFFu) << 16);                 // tensor_dim0[15:0] @ bits 63:48
    g1[2] = (int)(((tdim0 >> 16) & 0xFFFFu) |
                  ((tdim1 & 0xFFFFu) << 16));               // dim0[31:16] | dim1[15:0]
    g1[3] = (int)(((tdim1 >> 16) & 0xFFFFu) |
                  (tile0 << 16));                           // dim1[31:16] | tile_dim0
    g1[4] = (int)(tile1 & 0xFFFFu);                         // tile_dim1 (tile_dim2 = 0)
    g1[5] = (int)stride0;                                   // tensor_dim0_stride[31:0]
    g1[6] = 0;                                              // stride hi / dim1_stride lo
    g1[7] = 0;
    i32x4 z4 = {0, 0, 0, 0};
    i32x8 z8 = {0, 0, 0, 0, 0, 0, 0, 0};
    __builtin_amdgcn_tensor_load_to_lds(g0, g1, z4, z4, z8, 0);
}
#endif

// ---------------- kernel 1: convert x f32 -> bf16 ----------------
__global__ __launch_bounds__(256) void cvt_x_kernel(const float* __restrict__ x,
                                                    bf16* __restrict__ xb) {
    int i = (blockIdx.x * 256 + threadIdx.x) * 4;
    float4 v = *(const float4*)(x + i);
    xb[i + 0] = f2bf(v.x);
    xb[i + 1] = f2bf(v.y);
    xb[i + 2] = f2bf(v.z);
    xb[i + 3] = f2bf(v.w);
}

// ---------------- kernel 2: convert + transpose weights ----------------
__global__ __launch_bounds__(256) void cvt_wT_kernel(const float* __restrict__ w0,
                                                     const float* __restrict__ w1,
                                                     const float* __restrict__ w2,
                                                     const float* __restrict__ w3,
                                                     bf16* __restrict__ wt) {
    const float* w = (blockIdx.z == 0) ? w0 : (blockIdx.z == 1) ? w1
                   : (blockIdx.z == 2) ? w2 : w3;
    int idx = blockIdx.x * 256 + threadIdx.x;
    int n = idx >> 10;
    int k = idx & 1023;
    wt[(size_t)blockIdx.z * HIDDEN * HIDDEN + idx] = f2bf(w[(size_t)k * HIDDEN + n]);
}

// ---------------- kernel 3: bf16 tiled GEMM, TDM-staged, double-buffered ----
// C[M=4096, N=1024] = A[M,1024] * Bt[N,1024]^T + bias
// mode 0: bf16 out[h][s][d] (Q,K)   mode 1: bf16 out[h][d][s] (V)
// mode 2: f32 out[s][n] (final projection)
__global__ __launch_bounds__(256) void gemm_bf16_kernel(const bf16* __restrict__ A,
                                                        const bf16* __restrict__ Bt,
                                                        const float* __restrict__ bias,
                                                        void* __restrict__ Cout,
                                                        int mode) {
    __shared__ __attribute__((aligned(128))) bf16 As[2][128 * 64];
    __shared__ __attribute__((aligned(128))) bf16 Bs[2][128 * 64];

    const int tid  = threadIdx.x;
    const int lane = tid & 31;
    const int wave = tid >> 5;
    const int wm   = wave & 3;          // 4 waves along M (32 rows each)
    const int wn   = wave >> 2;         // 2 waves along N (64 cols each)
    const int l15  = lane & 15;
    const int lh   = lane >> 4;

    const int mb = blockIdx.x * 128;
    const int nb = blockIdx.y * 128;

    v8f acc[2][4] = {};

#ifdef HAVE_TDM
    // ---- prologue: TDM fetch of first K-tile into buffer 0 (wave 0 only;
    // tensor ops ignore EXEC and issue once per wave) ----
    if (wave == 0) {
        tdm_load_tile_2d(A  + (size_t)mb * HIDDEN, (unsigned)(uintptr_t)&As[0][0],
                         HIDDEN, S_LEN,  HIDDEN, 64, 128);
        tdm_load_tile_2d(Bt + (size_t)nb * HIDDEN, (unsigned)(uintptr_t)&Bs[0][0],
                         HIDDEN, HIDDEN, HIDDEN, 64, 128);
    }
#endif

    for (int kc = 0; kc < HIDDEN; kc += 64) {
        const int cur = (kc >> 6) & 1;

#ifdef HAVE_TDM
        // data for `cur` ready once wave0's TENSORcnt drains; barrier publishes
        if (wave == 0) __builtin_amdgcn_s_wait_tensorcnt(0);
        __syncthreads();
        // kick off DMA of the next K-tile into the other buffer (overlaps WMMA)
        if (wave == 0 && kc + 64 < HIDDEN) {
            tdm_load_tile_2d(A  + (size_t)mb * HIDDEN + kc + 64,
                             (unsigned)(uintptr_t)&As[1 - cur][0],
                             HIDDEN, S_LEN,  HIDDEN, 64, 128);
            tdm_load_tile_2d(Bt + (size_t)nb * HIDDEN + kc + 64,
                             (unsigned)(uintptr_t)&Bs[1 - cur][0],
                             HIDDEN, HIDDEN, HIDDEN, 64, 128);
        }
#else
        // fallback: cooperative staged copy (loads batched, then stores)
        {
            v8bf ra[2], rb[2];
            #pragma unroll
            for (int i = 0; i < 2; i++) {
                int idx = tid + i * 256;         // 512 chunks of 16B each array
                int row = idx >> 2;
                int c16 = (idx & 3) * 16;
                ra[i] = *(const v8bf*)(A  + (size_t)(mb + row) * HIDDEN + kc + c16);
                rb[i] = *(const v8bf*)(Bt + (size_t)(nb + row) * HIDDEN + kc + c16);
            }
            __syncthreads();
            #pragma unroll
            for (int i = 0; i < 2; i++) {
                int idx = tid + i * 256;
                int row = idx >> 2;
                int c16 = (idx & 3) * 16;
                *(v8bf*)(&As[cur][0] + row * 64 + c16) = ra[i];
                *(v8bf*)(&Bs[cur][0] + row * 64 + c16) = rb[i];
            }
            __syncthreads();
        }
#endif

        #pragma unroll
        for (int ks = 0; ks < 64; ks += 32) {
            v16bf a[2], b[4];
            #pragma unroll
            for (int tm = 0; tm < 2; tm++) {
                int m = wm * 32 + tm * 16 + l15;
                const bf16* p = &As[cur][0] + m * 64 + ks + lh * 16;  // contiguous k, matches B
                a[tm] = *(const v16bf*)p;
            }
            #pragma unroll
            for (int tn = 0; tn < 4; tn++) {
                int n = wn * 64 + tn * 16 + l15;
                b[tn] = *(const v16bf*)(&Bs[cur][0] + n * 64 + ks + lh * 16); // contiguous k, matches A
            }
            #pragma unroll
            for (int tm = 0; tm < 2; tm++)
                #pragma unroll
                for (int tn = 0; tn < 4; tn++)
                    acc[tm][tn] = __builtin_amdgcn_wmma_f32_16x16x32_bf16(
                        false, a[tm], false, b[tn], (short)0, acc[tm][tn], false, false);
        }
        __syncthreads();   // all waves done reading `cur` before it is re-filled
    }

    // epilogue: C layout -> lane holds (m = r + 8*lh, n = l15) per 16x16 tile
    #pragma unroll
    for (int tm = 0; tm < 2; tm++) {
        #pragma unroll
        for (int tn = 0; tn < 4; tn++) {
            int n = nb + wn * 64 + tn * 16 + l15;
            float bv = bias ? bias[n] : 0.0f;
            #pragma unroll
            for (int r = 0; r < 8; r++) {
                int m = mb + wm * 32 + tm * 16 + r + 8 * lh;
                float v = acc[tm][tn][r] + bv;
                if (mode == 0) {
                    ((bf16*)Cout)[(size_t)(n >> 6) * S_LEN * HDIM +
                                  (size_t)m * HDIM + (n & 63)] = f2bf(v);
                } else if (mode == 1) {
                    ((bf16*)Cout)[(size_t)(n >> 6) * HDIM * S_LEN +
                                  (size_t)(n & 63) * S_LEN + m] = f2bf(v);
                } else {
                    ((float*)Cout)[(size_t)m * HIDDEN + n] = v;
                }
            }
        }
    }
}

// ---------------- kernel 4: sliding-window flash attention ----------------
__global__ __launch_bounds__(128) void attn_kernel(const bf16* __restrict__ Q,
                                                   const bf16* __restrict__ K,
                                                   const bf16* __restrict__ VT,
                                                   bf16* __restrict__ AO) {
    __shared__ __attribute__((aligned(32))) bf16 Pbuf[4][16 * 32];

    const int tid  = threadIdx.x;
    const int lane = tid & 31;
    const int w    = tid >> 5;
    const int unit = blockIdx.x * 4 + w;       // 0..4095
    const int h    = unit >> 8;                // 256 q-blocks per head
    const int qb   = unit & 255;
    const int l15  = lane & 15;
    const int lh   = lane >> 4;

    const bf16* Qh  = Q  + (size_t)h * S_LEN * HDIM;
    const bf16* Kh  = K  + (size_t)h * S_LEN * HDIM;
    const bf16* VTh = VT + (size_t)h * HDIM * S_LEN;
    bf16* Ps = &Pbuf[w][0];

    const int qrow = qb * 16 + l15;
    v16bf aQ[2];
    #pragma unroll
    for (int t = 0; t < 2; t++) {
        const bf16* p = Qh + (size_t)qrow * HDIM + t * 32 + lh * 16;
        aQ[t] = *(const v16bf*)p;
    }

    float mrow[8], lrow[8];
    v8f O[4] = {};
    #pragma unroll
    for (int r = 0; r < 8; r++) { mrow[r] = -3.0e38f; lrow[r] = 0.0f; }

    const float sc = 0.125f;   // 1/sqrt(64)

    for (int c = 0; c < 9; c++) {
        int kb = qb * 16 - 128 + c * 32;

        // ---- scores: two 16-key tiles, K-dim = 64 split into 2 WMMAs ----
        v8f sv[2];
        #pragma unroll
        for (int kt = 0; kt < 2; kt++) {
            int key = kb + kt * 16 + l15;
            int kcl = min(max(key, 0), S_LEN - 1);
            v16bf b0 = *(const v16bf*)(Kh + (size_t)kcl * HDIM + lh * 16);
            v16bf b1 = *(const v16bf*)(Kh + (size_t)kcl * HDIM + 32 + lh * 16);
            v8f s = {};
            s = __builtin_amdgcn_wmma_f32_16x16x32_bf16(false, aQ[0], false, b0,
                                                        (short)0, s, false, false);
            s = __builtin_amdgcn_wmma_f32_16x16x32_bf16(false, aQ[1], false, b1,
                                                        (short)0, s, false, false);
            sv[kt] = s;
        }

        // ---- online softmax (per-row stats; rows live in half-waves) ----
        #pragma unroll
        for (int r = 0; r < 8; r++) {
            int qq = qb * 16 + r + 8 * lh;
            float s0 = sv[0][r] * sc, s1 = sv[1][r] * sc;
            int k0 = kb + l15, k1 = kb + 16 + l15;
            bool ok0 = (k0 >= 0) && (k0 < S_LEN) && (abs(qq - k0) <= WIN);
            bool ok1 = (k1 >= 0) && (k1 < S_LEN) && (abs(qq - k1) <= WIN);
            s0 = ok0 ? s0 : -3.0e38f;
            s1 = ok1 ? s1 : -3.0e38f;

            float rmax = redmax16(fmaxf(s0, s1));
            float mnew = fmaxf(mrow[r], rmax);
            float scale = __expf(mrow[r] - mnew);
            mrow[r] = mnew;

            float p0 = __expf(s0 - mnew);
            float p1 = __expf(s1 - mnew);
            lrow[r] = lrow[r] * scale + redsum16(p0 + p1);

            #pragma unroll
            for (int t = 0; t < 4; t++) O[t][r] *= scale;

            Ps[(r + 8 * lh) * 32 + l15]      = f2bf(p0);
            Ps[(r + 8 * lh) * 32 + 16 + l15] = f2bf(p1);
        }
        __syncthreads();   // uniform trip count across all waves in block

        // ---- re-read P in WMMA A-layout: row = l15, keys contiguous ----
        const bf16* pp = Ps + l15 * 32 + lh * 16;
        v16bf aP = *(const v16bf*)pp;

        // ---- O += P @ V  (V pre-transposed: contiguous 16-key reads) ----
        int ks = min(max(kb + lh * 16, 0), S_LEN - 16);
        #pragma unroll
        for (int t = 0; t < 4; t++) {
            int d = t * 16 + l15;
            v16bf bV = *(const v16bf*)(VTh + (size_t)d * S_LEN + ks);
            O[t] = __builtin_amdgcn_wmma_f32_16x16x32_bf16(false, aP, false, bV,
                                                           (short)0, O[t], false, false);
        }
        __syncthreads();
    }

    #pragma unroll
    for (int r = 0; r < 8; r++) {
        float inv = 1.0f / lrow[r];
        int srow = qb * 16 + r + 8 * lh;
        #pragma unroll
        for (int t = 0; t < 4; t++) {
            AO[(size_t)srow * HIDDEN + h * HDIM + t * 16 + l15] = f2bf(O[t][r] * inv);
        }
    }
}

// ---------------- launch ----------------
extern "C" void kernel_launch(void* const* d_in, const int* in_sizes, int n_in,
                              void* d_out, int out_size, void* d_ws, size_t ws_size,
                              hipStream_t stream) {
    const float* x  = (const float*)d_in[0];
    const float* wq = (const float*)d_in[1];
    const float* bq = (const float*)d_in[2];
    const float* wk = (const float*)d_in[3];
    const float* bk = (const float*)d_in[4];
    const float* wv = (const float*)d_in[5];
    const float* bv = (const float*)d_in[6];
    const float* wo = (const float*)d_in[7];
    const float* bo = (const float*)d_in[8];

    char* ws = (char*)d_ws;
    size_t oXB = 0;
    size_t oWT = oXB + (size_t)S_LEN * HIDDEN * 2;
    size_t oQ  = oWT + 4ull * HIDDEN * HIDDEN * 2;
    size_t oK  = oQ  + (size_t)HEADS * S_LEN * HDIM * 2;
    size_t oVT = oK  + (size_t)HEADS * S_LEN * HDIM * 2;
    size_t oAO = oVT + (size_t)HEADS * S_LEN * HDIM * 2;

    bf16* xb  = (bf16*)(ws + oXB);
    bf16* wt  = (bf16*)(ws + oWT);
    bf16* wtq = wt;
    bf16* wtk = wt + 1ull * HIDDEN * HIDDEN;
    bf16* wtv = wt + 2ull * HIDDEN * HIDDEN;
    bf16* wto = wt + 3ull * HIDDEN * HIDDEN;
    bf16* qb_ = (bf16*)(ws + oQ);
    bf16* kb_ = (bf16*)(ws + oK);
    bf16* vtb = (bf16*)(ws + oVT);
    bf16* ao  = (bf16*)(ws + oAO);

    cvt_x_kernel<<<(S_LEN * HIDDEN) / (256 * 4), 256, 0, stream>>>(x, xb);
    cvt_wT_kernel<<<dim3((HIDDEN * HIDDEN) / 256, 1, 4), 256, 0, stream>>>(wq, wk, wv, wo, wt);

    dim3 gg(S_LEN / 128, HIDDEN / 128);
    gemm_bf16_kernel<<<gg, 256, 0, stream>>>(xb, wtq, bq, qb_, 0);
    gemm_bf16_kernel<<<gg, 256, 0, stream>>>(xb, wtk, bk, kb_, 0);
    gemm_bf16_kernel<<<gg, 256, 0, stream>>>(xb, wtv, bv, vtb, 1);

    attn_kernel<<<(HEADS * (S_LEN / 16)) / 4, 128, 0, stream>>>(qb_, kb_, vtb, ao);

    gemm_bf16_kernel<<<gg, 256, 0, stream>>>(ao, wto, bo, (float*)d_out, 2);
}
